// MGATModel_36971078484254
// MI455X (gfx1250) — matrix operand order repaired and checked by reading, not compiled, not run on hardware
//
#include <hip/hip_runtime.h>
#include <hip/hip_bf16.h>

typedef __attribute__((ext_vector_type(16))) _Float16 v16h;
typedef __attribute__((ext_vector_type(8)))  _Float16 v8h;
typedef __attribute__((ext_vector_type(8)))  float    v8f;

#define BQ 8
#define NQ 1024
#define HQ 128
#define NHQ 4
#define DHQ 32

union AB16 { v16h v; v8h h[2]; };

// ---------------------------------------------------------------------------
// Generic WMMA GEMM: C[M,Nn] = A[M,K] @ Bw[Nn,K]^T (+bias) (+=C if accumulate)
// 16x32 output tile per wave (two 16x16 WMMA chains sharing one A fetch).
// Optional f16 mirror Cf16[M,Nn]; optional head-transposed mirror
// CT[b][head][d][n] (for M = B*N, Nn = 128, head = col/32, d = col%32).
// Requires Nn % 32 == 0, K % 32 == 0.
// ---------------------------------------------------------------------------
__global__ __launch_bounds__(128) void gemm_wmma_kernel(
    const _Float16* __restrict__ A, const _Float16* __restrict__ Bw,
    const float* __restrict__ bias, float* __restrict__ C, int accumulate,
    _Float16* __restrict__ Cf16, _Float16* __restrict__ CT,
    int M, int Nn, int K)
{
  int wave = threadIdx.x >> 5;
  int lane = threadIdx.x & 31;
  int tiles_n = Nn >> 5;                 // 32-wide N tiles
  int tile = blockIdx.x * 4 + wave;
  int total = (M >> 4) * tiles_n;
  if (tile >= total) return;
  int tm = tile / tiles_n, tn = tile - tm * tiles_n;
  int mbase = tm << 4, nbase = tn << 5;
  int row = lane & 15;
  int hl  = lane >> 4;                   // lane half: 0 or 1

  const _Float16* arow  = A  + (size_t)(mbase + row) * K + hl * 8;
  const _Float16* brow0 = Bw + (size_t)(nbase + row) * K + hl * 8;
  const _Float16* brow1 = brow0 + (size_t)16 * K;

  v8f acc0 = {}, acc1 = {};
  for (int k = 0; k < K; k += 32) {
    AB16 a, b0, b1;
    a.h[0]  = *(const v8h*)(arow + k);
    a.h[1]  = *(const v8h*)(arow + k + 16);
    b0.h[0] = *(const v8h*)(brow0 + k);
    b0.h[1] = *(const v8h*)(brow0 + k + 16);
    b1.h[0] = *(const v8h*)(brow1 + k);
    b1.h[1] = *(const v8h*)(brow1 + k + 16);
    acc0 = __builtin_amdgcn_wmma_f32_16x16x32_f16(
        false, a.v, false, b0.v, (short)0, acc0, false, false);
    acc1 = __builtin_amdgcn_wmma_f32_16x16x32_f16(
        false, a.v, false, b1.v, (short)0, acc1, false, false);
  }

#pragma unroll
  for (int t = 0; t < 2; ++t) {
    int n = nbase + t * 16 + (lane & 15);
    float bval = bias ? bias[n] : 0.0f;
    v8f acc = t ? acc1 : acc0;
#pragma unroll
    for (int r = 0; r < 8; ++r) {
      int m = mbase + r + hl * 8;
      float v = acc[r] + bval;
      size_t idx = (size_t)m * Nn + n;
      if (C) {
        if (accumulate) v += C[idx];
        C[idx] = v;
      }
      if (Cf16) Cf16[idx] = (_Float16)v;
      if (CT) {
        int bb = m >> 10, nv = m & 1023, hh = n >> 5, dd = n & 31;
        CT[(((size_t)bb * NHQ + hh) * DHQ + dd) * NQ + nv] = (_Float16)v;
      }
    }
  }
}

// ---------------------------------------------------------------------------
// Flash-style masked self-attention (one GAT half), WMMA f16.
// Wh  : [B, N, 128] f16   (Q rows, K rows; head h = cols h*32..h*32+31)
// WhT : [B, 4, 32, N] f16 (V, transposed for contiguous B-operand loads)
// mask: [N, N] f16 (0 / -inf)
// Hout: [B, N, 256] f16, write elu(attn@V) at columns col_off + h*32 ...
// One wave per (b, head, 16-row query tile); 32 key blocks of 32.
// Row-sum kept as per-lane partials; reduced once in the epilogue.
// ---------------------------------------------------------------------------
__global__ __launch_bounds__(128) void attn_wmma_kernel(
    const _Float16* __restrict__ Wh, const _Float16* __restrict__ WhT,
    const _Float16* __restrict__ mask, _Float16* __restrict__ Hout, int col_off)
{
  __shared__ __align__(32) _Float16 pst[4][16 * 32];

  int wave = threadIdx.x >> 5;
  int lane = threadIdx.x & 31;
  int gid  = blockIdx.x * 4 + wave;      // 0 .. 2047
  int tile = gid & 63;                   // query row tile
  int head = (gid >> 6) & 3;
  int b    = gid >> 8;
  int col  = lane & 15;
  int hl   = lane >> 4;
  int rowg = tile * 16 + col;            // global query row for A loads

  // Q as A-operand (persistent)
  const _Float16* qp = Wh + ((size_t)(b * NQ + rowg) * HQ) + head * DHQ + hl * 8;
  AB16 qa;
  qa.h[0] = *(const v8h*)(qp);
  qa.h[1] = *(const v8h*)(qp + 16);

  v8f o0 = {}, o1 = {};
  float mrow[8], lrow[8];
#pragma unroll
  for (int r = 0; r < 8; ++r) { mrow[r] = -3.0e38f; lrow[r] = 0.0f; }

  const float scale = 0.17677669529663687f;   // 1/sqrt(32)
  _Float16* sm = &pst[wave][0];
  const _Float16* vbase = WhT + ((size_t)(b * NHQ + head) * DHQ) * NQ;

  for (int j = 0; j < NQ; j += 32) {
    // ---- scores: two 16x16 wmma tiles over keys j..j+31 ----
    AB16 kb0, kb1;
    const _Float16* kp0 = Wh + ((size_t)(b * NQ + j + col) * HQ) + head * DHQ + hl * 8;
    const _Float16* kp1 = kp0 + (size_t)16 * HQ;
    kb0.h[0] = *(const v8h*)kp0; kb0.h[1] = *(const v8h*)(kp0 + 16);
    kb1.h[0] = *(const v8h*)kp1; kb1.h[1] = *(const v8h*)(kp1 + 16);

    v8f zero = {};
    v8f s0 = __builtin_amdgcn_wmma_f32_16x16x32_f16(
        false, qa.v, false, kb0.v, (short)0, zero, false, false);
    v8f s1 = __builtin_amdgcn_wmma_f32_16x16x32_f16(
        false, qa.v, false, kb1.v, (short)0, zero, false, false);

    // ---- online softmax per row (row r in C-layout, 16 lanes/half) ----
#pragma unroll
    for (int r = 0; r < 8; ++r) {
      int mg = tile * 16 + r + hl * 8;
      float v0 = s0[r] * scale + (float)mask[(size_t)mg * NQ + j + col];
      float v1 = s1[r] * scale + (float)mask[(size_t)mg * NQ + j + 16 + col];
      float mx = fmaxf(v0, v1);
      mx = fmaxf(mx, __shfl_xor(mx, 1, 32));
      mx = fmaxf(mx, __shfl_xor(mx, 2, 32));
      mx = fmaxf(mx, __shfl_xor(mx, 4, 32));
      mx = fmaxf(mx, __shfl_xor(mx, 8, 32));
      float mnew = fmaxf(mrow[r], mx);          // never -inf: mrow >= -3e38
      float alpha = __expf(mrow[r] - mnew);
      float p0 = __expf(v0 - mnew);
      float p1 = __expf(v1 - mnew);
      // per-lane partial row sum; cross-lane reduce deferred to epilogue
      lrow[r] = lrow[r] * alpha + (p0 + p1);
      mrow[r] = mnew;
      o0[r] = o0[r] * alpha;
      o1[r] = o1[r] * alpha;
      // stage P (C layout -> LDS row-major tile) for A-operand re-load
      int rloc = r + hl * 8;
      sm[rloc * 32 + col]      = (_Float16)p0;
      sm[rloc * 32 + col + 16] = (_Float16)p1;
    }
    asm volatile("s_wait_dscnt 0" ::: "memory");

    // ---- P as A-operand ----
    AB16 pa;
    pa.h[0] = *(const v8h*)(sm + col * 32 + hl * 16);
    pa.h[1] = *(const v8h*)(sm + col * 32 + hl * 16 + 8);

    // ---- V as B-operands (contiguous from WhT) ----
    AB16 vb0, vb1;
    const _Float16* vp0 = vbase + (size_t)col * NQ + j + hl * 16;
    const _Float16* vp1 = vbase + (size_t)(16 + col) * NQ + j + hl * 16;
    vb0.h[0] = *(const v8h*)vp0; vb0.h[1] = *(const v8h*)(vp0 + 8);
    vb1.h[0] = *(const v8h*)vp1; vb1.h[1] = *(const v8h*)(vp1 + 8);

    o0 = __builtin_amdgcn_wmma_f32_16x16x32_f16(
        false, pa.v, false, vb0.v, (short)0, o0, false, false);
    o1 = __builtin_amdgcn_wmma_f32_16x16x32_f16(
        false, pa.v, false, vb1.v, (short)0, o1, false, false);
  }

  // ---- epilogue: reduce row sums, normalize, ELU, store f16 ----
#pragma unroll
  for (int r = 0; r < 8; ++r) {
    float ls = lrow[r];
    ls += __shfl_xor(ls, 1, 32);
    ls += __shfl_xor(ls, 2, 32);
    ls += __shfl_xor(ls, 4, 32);
    ls += __shfl_xor(ls, 8, 32);
    int mg = tile * 16 + r + hl * 8;
    float inv = 1.0f / ls;
    float e0 = o0[r] * inv, e1 = o1[r] * inv;
    e0 = (e0 > 0.0f) ? e0 : (__expf(e0) - 1.0f);
    e1 = (e1 > 0.0f) ? e1 : (__expf(e1) - 1.0f);
    size_t base = ((size_t)b * NQ + mg) * 256 + col_off + head * DHQ;
    Hout[base + col]      = (_Float16)e0;
    Hout[base + 16 + col] = (_Float16)e1;
  }
}

// ---------------------------------------------------------------------------
// Stim encoder: Linear(3072->256) -> LayerNorm -> GELU(exact) -> Linear(256->128)
// One block per batch row.
// ---------------------------------------------------------------------------
__global__ __launch_bounds__(256) void encoder_kernel(
    const float* __restrict__ stim, const float* __restrict__ W1,
    const float* __restrict__ b1, const float* __restrict__ lng,
    const float* __restrict__ lnb, const float* __restrict__ W2,
    const float* __restrict__ b2, float* __restrict__ g)
{
  __shared__ float zsh[256];
  __shared__ float ws[8], wss[8];
  int b = blockIdx.x, t = threadIdx.x;
  const float* srow = stim + (size_t)b * 3072;
  const float* wrow = W1 + (size_t)t * 3072;
  float acc = b1[t];
  for (int k = 0; k < 3072; k += 4)
    acc += srow[k] * wrow[k] + srow[k + 1] * wrow[k + 1] +
           srow[k + 2] * wrow[k + 2] + srow[k + 3] * wrow[k + 3];

  float s = acc, ss = acc * acc;
  for (int off = 1; off < 32; off <<= 1) {
    s  += __shfl_xor(s,  off, 32);
    ss += __shfl_xor(ss, off, 32);
  }
  if ((t & 31) == 0) { ws[t >> 5] = s; wss[t >> 5] = ss; }
  __syncthreads();
  float S = 0.0f, SS = 0.0f;
  for (int i = 0; i < 8; ++i) { S += ws[i]; SS += wss[i]; }
  float mu  = S * (1.0f / 256.0f);
  float var = SS * (1.0f / 256.0f) - mu * mu;
  float zn  = (acc - mu) * rsqrtf(var + 1e-5f) * lng[t] + lnb[t];
  zsh[t] = 0.5f * zn * (1.0f + erff(zn * 0.70710678118654752f));
  __syncthreads();
  if (t < 128) {
    const float* w2 = W2 + (size_t)t * 256;
    float a2 = b2[t];
    for (int k = 0; k < 256; ++k) a2 += zsh[k] * w2[k];
    g[b * 128 + t] = a2;
  }
}

__global__ void buildx_kernel(const float* __restrict__ g,
                              const float* __restrict__ vox,
                              _Float16* __restrict__ xo)
{
  int i = blockIdx.x * 256 + threadIdx.x;
  if (i >= BQ * NQ * HQ) return;
  int k = i & 127, n = (i >> 7) & 1023, b = i >> 17;
  xo[i] = (_Float16)(g[b * 128 + k] + vox[n * 128 + k]);
}

__global__ void cvt_f16_kernel(const float* __restrict__ in,
                               _Float16* __restrict__ out, int n)
{
  int i = blockIdx.x * 256 + threadIdx.x;
  if (i < n) out[i] = (_Float16)in[i];
}

__global__ void readout_kernel(const float* __restrict__ x,
                               const float* __restrict__ wro,
                               const float* __restrict__ bro,
                               float* __restrict__ pred)
{
  int i = blockIdx.x * 256 + threadIdx.x;
  if (i >= BQ * NQ) return;
  const float* xr = x + (size_t)i * HQ;
  float a = bro[0];
  for (int k = 0; k < HQ; ++k) a += xr[k] * wro[k];
  pred[i] = a;
}

// ---------------------------------------------------------------------------
extern "C" void kernel_launch(void* const* d_in, const int* in_sizes, int n_in,
                              void* d_out, int out_size, void* d_ws, size_t ws_size,
                              hipStream_t stream) {
  (void)in_sizes; (void)n_in; (void)out_size; (void)ws_size;
  const float* stim    = (const float*)d_in[0];
  const float* W_enc1  = (const float*)d_in[1];
  const float* b_enc1  = (const float*)d_in[2];
  const float* ln_g    = (const float*)d_in[3];
  const float* ln_b    = (const float*)d_in[4];
  const float* W_enc2  = (const float*)d_in[5];
  const float* b_enc2  = (const float*)d_in[6];
  const float* voxel   = (const float*)d_in[7];
  const float* W_func1 = (const float*)d_in[8];
  const float* W_roi1  = (const float*)d_in[9];
  const float* W_func2 = (const float*)d_in[10];
  const float* W_roi2  = (const float*)d_in[11];
  const float* W_fu1   = (const float*)d_in[12];
  const float* b_fu1   = (const float*)d_in[13];
  const float* W_fu2   = (const float*)d_in[14];
  const float* b_fu2   = (const float*)d_in[15];
  const float* W_sk1   = (const float*)d_in[16];
  const float* b_sk1   = (const float*)d_in[17];
  const float* W_sk2   = (const float*)d_in[18];
  const float* b_sk2   = (const float*)d_in[19];
  const float* W_ro    = (const float*)d_in[20];
  const float* b_ro    = (const float*)d_in[21];
  const float* mask_f  = (const float*)d_in[22];
  const float* mask_r  = (const float*)d_in[23];
  float* pred = (float*)d_out;

  // workspace carve-out
  char* ws = (char*)d_ws;
  size_t off = 0;
  auto carve = [&](size_t bytes) -> void* {
    off = (off + 255) & ~(size_t)255;
    void* p = ws + off;
    off += bytes;
    return p;
  };
  const size_t MN = (size_t)BQ * NQ;               // 8192
  float*     g_f32  = (float*)    carve(BQ * HQ * sizeof(float));
  _Float16*  x_f16  = (_Float16*) carve(MN * HQ * 2);
  _Float16*  wh_f   = (_Float16*) carve(MN * HQ * 2);
  _Float16*  wh_r   = (_Float16*) carve(MN * HQ * 2);
  _Float16*  whT_f  = (_Float16*) carve(MN * HQ * 2);
  _Float16*  whT_r  = (_Float16*) carve(MN * HQ * 2);
  _Float16*  hcat   = (_Float16*) carve(MN * 256 * 2);
  float*     x2_f32 = (float*)    carve(MN * HQ * sizeof(float));
  _Float16*  x2_f16 = (_Float16*) carve(MN * HQ * 2);
  float*     x3_f32 = (float*)    carve(MN * HQ * sizeof(float));
  _Float16*  mfh    = (_Float16*) carve((size_t)NQ * NQ * 2);
  _Float16*  mrh    = (_Float16*) carve((size_t)NQ * NQ * 2);
  _Float16*  wf1h   = (_Float16*) carve(128 * 128 * 2);
  _Float16*  wr1h   = (_Float16*) carve(128 * 128 * 2);
  _Float16*  wf2h   = (_Float16*) carve(128 * 128 * 2);
  _Float16*  wr2h   = (_Float16*) carve(128 * 128 * 2);
  _Float16*  wfu1h  = (_Float16*) carve(128 * 256 * 2);
  _Float16*  wfu2h  = (_Float16*) carve(128 * 256 * 2);
  _Float16*  wsk1h  = (_Float16*) carve(128 * 128 * 2);
  _Float16*  wsk2h  = (_Float16*) carve(128 * 128 * 2);

  auto cvt = [&](const float* src, _Float16* dst, int n) {
    cvt_f16_kernel<<<(n + 255) / 256, 256, 0, stream>>>(src, dst, n);
  };
  cvt(W_func1, wf1h, 128 * 128);
  cvt(W_roi1,  wr1h, 128 * 128);
  cvt(W_func2, wf2h, 128 * 128);
  cvt(W_roi2,  wr2h, 128 * 128);
  cvt(W_fu1,   wfu1h, 128 * 256);
  cvt(W_fu2,   wfu2h, 128 * 256);
  cvt(W_sk1,   wsk1h, 128 * 128);
  cvt(W_sk2,   wsk2h, 128 * 128);
  cvt(mask_f,  mfh, NQ * NQ);
  cvt(mask_r,  mrh, NQ * NQ);

  // encoder + x = g + voxel_emb
  encoder_kernel<<<BQ, 256, 0, stream>>>(stim, W_enc1, b_enc1, ln_g, ln_b,
                                         W_enc2, b_enc2, g_f32);
  buildx_kernel<<<(BQ * NQ * HQ + 255) / 256, 256, 0, stream>>>(g_f32, voxel, x_f16);

  auto gemm = [&](const _Float16* A, const _Float16* Bw, const float* bias,
                  float* C, int acc, _Float16* Cf16, _Float16* CT,
                  int M, int Nn, int K) {
    int tiles = (M / 16) * (Nn / 32);
    gemm_wmma_kernel<<<(tiles + 3) / 4, 128, 0, stream>>>(A, Bw, bias, C, acc,
                                                          Cf16, CT, M, Nn, K);
  };
  auto attn = [&](const _Float16* Wh, const _Float16* WhT, const _Float16* mask,
                  int col_off) {
    attn_wmma_kernel<<<(BQ * NHQ * (NQ / 16)) / 4, 128, 0, stream>>>(
        Wh, WhT, mask, hcat, col_off);
  };

  // ---- GAT layer 1 ----
  gemm(x_f16, wf1h, nullptr, nullptr, 0, wh_f, whT_f, (int)MN, HQ, HQ);
  gemm(x_f16, wr1h, nullptr, nullptr, 0, wh_r, whT_r, (int)MN, HQ, HQ);
  attn(wh_f, whT_f, mfh, 0);
  attn(wh_r, whT_r, mrh, 128);
  gemm(hcat,  wfu1h, b_fu1, x2_f32, 0, nullptr, nullptr, (int)MN, HQ, 256);
  gemm(x_f16, wsk1h, b_sk1, x2_f32, 1, x2_f16, nullptr, (int)MN, HQ, HQ);

  // ---- GAT layer 2 ----
  gemm(x2_f16, wf2h, nullptr, nullptr, 0, wh_f, whT_f, (int)MN, HQ, HQ);
  gemm(x2_f16, wr2h, nullptr, nullptr, 0, wh_r, whT_r, (int)MN, HQ, HQ);
  attn(wh_f, whT_f, mfh, 0);
  attn(wh_r, whT_r, mrh, 128);
  gemm(hcat,   wfu2h, b_fu2, x3_f32, 0, nullptr, nullptr, (int)MN, HQ, 256);
  gemm(x2_f16, wsk2h, b_sk2, x3_f32, 1, nullptr, nullptr, (int)MN, HQ, HQ);

  // ---- readout ----
  readout_kernel<<<(BQ * NQ + 255) / 256, 256, 0, stream>>>(x3_f32, W_ro, b_ro, pred);
}
